// LeanableUpsampler_82282983457321
// MI455X (gfx1250) — matrix operand order repaired and checked by reading, not compiled, not run on hardware
//
#include <hip/hip_runtime.h>

typedef _Float16 v16h __attribute__((ext_vector_type(16)));
typedef _Float16 v8h  __attribute__((ext_vector_type(8)));
typedef float    v8f  __attribute__((ext_vector_type(8)));
typedef float    v4f  __attribute__((ext_vector_type(4)));
typedef int      v4i  __attribute__((__vector_size__(16)));

constexpr int B_ = 16;
constexpr int K_ = 512;
constexpr int D_ = 256;
constexpr int T_ = 4096;
constexpr int WP = 520;  // padded K pitch (halves): 1040B rows -> 4-bank stagger, b128 conflict-free

#if defined(__has_builtin)
#if __has_builtin(__builtin_amdgcn_global_load_async_to_lds_b128)
#define HAS_ASYNC_LDS 1
#endif
#endif

// ---------------------------------------------------------------------------
// Kernel 1: per-batch inclusive scan of durations -> phoneme centers (B x K)
// ---------------------------------------------------------------------------
__global__ __launch_bounds__(512) void centers_kernel(const float* __restrict__ dur,
                                                      float* __restrict__ centers) {
  __shared__ float s[K_];
  const int b = blockIdx.x;
  const int k = threadIdx.x;
  const float d = dur[b * K_ + k];
  s[k] = d;
  __syncthreads();
#pragma unroll
  for (int off = 1; off < K_; off <<= 1) {
    const float t = (k >= off) ? s[k - off] : 0.f;
    __syncthreads();
    s[k] += t;
    __syncthreads();
  }
  centers[b * K_ + k] = s[k] - 0.5f * d;  // cumsum - dur/2
}

// ---------------------------------------------------------------------------
// Kernel 2: phoneme f32 -> f16 (once, into L2-resident ws; 4 MB)
// NT load (read-once stream), RT store (re-read 64x by the main kernel).
// ---------------------------------------------------------------------------
__global__ __launch_bounds__(256) void cvt_phoneme_kernel(const float* __restrict__ src,
                                                          _Float16* __restrict__ dst) {
  const size_t i = ((size_t)blockIdx.x * 256 + threadIdx.x) * 8;
  const v4f f0 = __builtin_nontemporal_load((const v4f*)(src + i));
  const v4f f1 = __builtin_nontemporal_load((const v4f*)(src + i + 4));
  v8h h;
#pragma unroll
  for (int j = 0; j < 4; ++j) {
    h[j]     = (_Float16)f0[j];
    h[j + 4] = (_Float16)f1[j];
  }
  *(v8h*)(dst + i) = h;
}

__device__ __forceinline__ v16h cat16(v8h lo, v8h hi) {
  return __builtin_shufflevector(lo, hi, 0, 1, 2, 3, 4, 5, 6, 7, 8, 9, 10, 11, 12, 13, 14, 15);
}

// ---------------------------------------------------------------------------
// Kernel 3: fused softmax-alignment + WMMA GEMM + frame residual
// block: 256 threads (8 wave32); tile: 64 t-rows x 64 d-cols; all K=512
// ---------------------------------------------------------------------------
__global__ __launch_bounds__(256) void upsampler_kernel(const float* __restrict__ centers_g,
                                                        const _Float16* __restrict__ phonemeH,
                                                        const float* __restrict__ frame,
                                                        float* __restrict__ out) {
  extern __shared__ char smem[];
  float*    cLds = (float*)smem;                                // 512 f32   (2 KB)
  _Float16* wLds = (_Float16*)(smem + 2048);                    // 64 x WP f16 (softmax weights)
  _Float16* pLds = (_Float16*)(smem + 2048 + 64 * WP * 2);      // 64 x WP f16 (phoneme slice)

  const int tid  = threadIdx.x;
  const int b    = blockIdx.z;
  const int dblk = blockIdx.y * 64;
  const int tblk = blockIdx.x * 64;

  // Prefetch the frame tile this block will read in the epilogue (global_prefetch_b8).
  {
    const int fr = tid & 63;
    const int fs = tid >> 6;
    __builtin_prefetch(frame + ((size_t)(b * D_ + dblk + fr)) * T_ + tblk + fs * 16, 0, 0);
  }

  // Phase 1a: centers -> LDS (2 per thread).
  cLds[tid]       = centers_g[b * K_ + tid];
  cLds[tid + 256] = centers_g[b * K_ + tid + 256];

  // Phase 1b: stage f16 phoneme slice ws -> LDS (async memory->LDS path, ASYNCcnt;
  // each thread copies 16 x 16B chunks of its (row, k-quarter) segment).
  {
    const int prow  = tid >> 2;
    const int kbase = (tid & 3) * 128;
    const _Float16* src = phonemeH + ((size_t)(b * D_ + dblk + prow)) * K_ + kbase;
    _Float16*       dst = pLds + prow * WP + kbase;
#ifdef HAS_ASYNC_LDS
    typedef __attribute__((address_space(1))) v4i g_v4i;
    typedef __attribute__((address_space(3))) v4i l_v4i;
#pragma unroll
    for (int c = 0; c < 16; ++c) {
      __builtin_amdgcn_global_load_async_to_lds_b128(
          (g_v4i*)(src + c * 8),   // addrspacecast generic -> global
          (l_v4i*)(dst + c * 8),   // addrspacecast generic -> LDS
          0, 0);
    }
#if __has_builtin(__builtin_amdgcn_s_wait_asynccnt)
    __builtin_amdgcn_s_wait_asynccnt(0);
#else
    asm volatile("s_wait_asynccnt 0x0" ::: "memory");
#endif
#else
#pragma unroll
    for (int c = 0; c < 16; ++c) {
      *(v8h*)(dst + c * 8) = *(const v8h*)(src + c * 8);
    }
#endif
  }
  __syncthreads();

  // Phase 2: softmax over K per t-row. 4 lanes per row (lane groups aligned in wave32),
  // shfl_xor reductions for row max and row sum; normalized f16 weights -> wLds.
  {
    const int   row   = tid >> 2;
    const int   kbase = (tid & 3) * 128;
    const float tc    = (float)(tblk + row) + 0.5f;

    float m = -3.0e38f;
#pragma unroll 8
    for (int kk = 0; kk < 128; ++kk) {
      const float dd = tc - cLds[kbase + kk];
      m = fmaxf(m, -dd * dd);
    }
    m = fmaxf(m, __shfl_xor(m, 1));
    m = fmaxf(m, __shfl_xor(m, 2));

    float s = 0.f;
    for (int kk = 0; kk < 128; kk += 8) {
      v8h p;
#pragma unroll
      for (int i = 0; i < 8; ++i) {
        const float dd = tc - cLds[kbase + kk + i];
        const float e  = __expf(-dd * dd - m);
        s += e;
        p[i] = (_Float16)e;
      }
      *(v8h*)(wLds + row * WP + kbase + kk) = p;
    }
    s += __shfl_xor(s, 1);
    s += __shfl_xor(s, 2);
    const float inv = 1.f / s;

    for (int kk = 0; kk < 128; kk += 8) {
      v8h p = *(v8h*)(wLds + row * WP + kbase + kk);
#pragma unroll
      for (int i = 0; i < 8; ++i) p[i] = (_Float16)((float)p[i] * inv);
      *(v8h*)(wLds + row * WP + kbase + kk) = p;
    }
  }
  __syncthreads();

  // Phase 3: WMMA. wave w -> row-tile rt = w&3, col-tiles {ct0, ct0+1}, ct0 = (w>>2)*2.
  const int lane  = tid & 31;
  const int wv    = tid >> 5;
  const int rt    = wv & 3;
  const int ct0   = (wv >> 2) * 2;
  const int lhalf = lane & 15;
  const int hi    = lane >> 4;  // 0: lanes 0-15, 1: lanes 16-31

  v8f acc0 = {0.f, 0.f, 0.f, 0.f, 0.f, 0.f, 0.f, 0.f};
  v8f acc1 = {0.f, 0.f, 0.f, 0.f, 0.f, 0.f, 0.f, 0.f};

  const _Float16* arow  = wLds + (rt * 16 + lhalf) * WP;          // A row = t
  const _Float16* brow0 = pLds + (ct0 * 16 + lhalf) * WP;         // B col = d
  const _Float16* brow1 = pLds + ((ct0 + 1) * 16 + lhalf) * WP;

#pragma unroll 4
  for (int ks = 0; ks < 16; ++ks) {
    const int kk = ks * 32;
    // A (16x32 f16): lanes 0-15 hold K kk+0..15; lanes 16-31 hold K kk+16..31.
    const int ka = kk + (hi ? 16 : 0);
    const v16h a = cat16(*(const v8h*)(arow + ka), *(const v8h*)(arow + ka + 8));
    // B (32x16 f16): lanes 0-15 hold K kk+0..15; lanes 16-31 hold K kk+16..31.
    const int kb = kk + (hi ? 16 : 0);
    const v16h b0 = cat16(*(const v8h*)(brow0 + kb), *(const v8h*)(brow0 + kb + 8));
    const v16h b1 = cat16(*(const v8h*)(brow1 + kb), *(const v8h*)(brow1 + kb + 8));

    acc0 = __builtin_amdgcn_wmma_f32_16x16x32_f16(false, a, false, b0, (short)0, acc0, false, false);
    acc1 = __builtin_amdgcn_wmma_f32_16x16x32_f16(false, a, false, b1, (short)0, acc1, false, false);
  }

  // Phase 4: epilogue. C/D layout: lane = d column (N=lane%16), VGPR j = t row (M=j+8*hi).
  // Each lane owns 8 consecutive t values of out[b][d][...]; frame/out are read/written
  // exactly once -> non-temporal to keep L2 for the reused phoneme data.
  const int trow = tblk + rt * 16 + hi * 8;
  {
    const size_t o = ((size_t)(b * D_ + dblk + ct0 * 16 + lhalf)) * T_ + trow;
    v4f f0 = __builtin_nontemporal_load((const v4f*)(frame + o));
    v4f f1 = __builtin_nontemporal_load((const v4f*)(frame + o + 4));
    f0 += __builtin_shufflevector(acc0, acc0, 0, 1, 2, 3);
    f1 += __builtin_shufflevector(acc0, acc0, 4, 5, 6, 7);
    __builtin_nontemporal_store(f0, (v4f*)(out + o));
    __builtin_nontemporal_store(f1, (v4f*)(out + o + 4));
  }
  {
    const size_t o = ((size_t)(b * D_ + dblk + (ct0 + 1) * 16 + lhalf)) * T_ + trow;
    v4f f0 = __builtin_nontemporal_load((const v4f*)(frame + o));
    v4f f1 = __builtin_nontemporal_load((const v4f*)(frame + o + 4));
    f0 += __builtin_shufflevector(acc1, acc1, 0, 1, 2, 3);
    f1 += __builtin_shufflevector(acc1, acc1, 4, 5, 6, 7);
    __builtin_nontemporal_store(f0, (v4f*)(out + o));
    __builtin_nontemporal_store(f1, (v4f*)(out + o + 4));
  }
}

extern "C" void kernel_launch(void* const* d_in, const int* in_sizes, int n_in,
                              void* d_out, int out_size, void* d_ws, size_t ws_size,
                              hipStream_t stream) {
  const float* durations = (const float*)d_in[0];  // (B, K)
  const float* phoneme   = (const float*)d_in[1];  // (B, D, K)
  const float* frame     = (const float*)d_in[2];  // (B, D, T)
  float*       out       = (float*)d_out;          // (B, D, T)

  // ws layout: [0, 32KB) centers f32; [32KB, 32KB+4MB) phoneme f16
  float*    centers  = (float*)d_ws;
  _Float16* phonemeH = (_Float16*)((char*)d_ws + (size_t)B_ * K_ * sizeof(float));

  centers_kernel<<<dim3(B_), dim3(K_), 0, stream>>>(durations, centers);

  const int nPh = B_ * D_ * K_;  // 2,097,152 elements; 8 per thread
  cvt_phoneme_kernel<<<dim3(nPh / (256 * 8)), dim3(256), 0, stream>>>(phoneme, phonemeH);

  const size_t ldsBytes = 2048 + 2 * (size_t)64 * WP * sizeof(_Float16);  // 135,168 B
  upsampler_kernel<<<dim3(T_ / 64, D_ / 64, B_), dim3(256), ldsBytes, stream>>>(
      centers, phonemeH, frame, out);
}
